// BVNet_70738111365458
// MI455X (gfx1250) — matrix operand order repaired and checked by reading, not compiled, not run on hardware
//
#include <hip/hip_runtime.h>
#include <hip/hip_bf16.h>
#include <float.h>

// ---------------------------------------------------------------------------
// GATv2 x2 + JK-cat readout for MI455X (gfx1250, wave32, WMMA).
// Node transforms: v_wmma_f32_16x16x32_bf16 with fragments pre-packed in the
// contiguous-k layout (k = 16*half + i) (A staged in LDS as b128-loadable v16bf, B packed
// once in global and served from L2). Edge softmax/aggregate kernels are
// L2-resident gather/scatter (xl/xr = 25.6MB each, fits the 192MB L2).
// ---------------------------------------------------------------------------

typedef __attribute__((ext_vector_type(16))) __bf16   v16bf;
typedef __attribute__((ext_vector_type(8)))  float    v8f;
typedef __attribute__((ext_vector_type(8)))  unsigned v8u;

#define NEG_SLOPE 0.2f
#define EPS_DEN   1e-16f

// monotonic float -> uint mapping for atomic segment-max
__device__ __forceinline__ unsigned fkey(float f) {
    unsigned b = __float_as_uint(f);
    return (b & 0x80000000u) ? ~b : (b | 0x80000000u);
}
__device__ __forceinline__ float funkey(unsigned k) {
    unsigned b = (k & 0x80000000u) ? (k & 0x7FFFFFFFu) : ~k;
    return __uint_as_float(b);
}
#define KEY_NEGMAX 0x00800000u  /* fkey(-FLT_MAX) */

__device__ __forceinline__ unsigned pack_bf16x2(float lo, float hi) {
    unsigned short l = __builtin_bit_cast(unsigned short, (__bf16)lo);
    unsigned short h = __builtin_bit_cast(unsigned short, (__bf16)hi);
    return ((unsigned)h << 16) | (unsigned)l;
}

// ---------------------------------------------------------------------------
__global__ void fill_u32(unsigned* __restrict__ p, unsigned v, size_t n) {
    size_t i = (size_t)blockIdx.x * blockDim.x + threadIdx.x;
    if (i < n) p[i] = v;
}

// ---------------------------------------------------------------------------
// Pack W[64][64] f32 (row-major KxN) into B-fragment layout:
// Wp dword index = ((s*4 + w)*32 + lane)*8 + v
//   s = kstep (0,1), w = column tile, lane: half=lane>>4 col=16w+(lane&15)
//   dword v holds bf16 pair for k = 32s + 16*half + 2v (+1)
// 2048 dwords total (8KB) -> lives in L2, broadcast to all GEMM blocks.
// ---------------------------------------------------------------------------
__global__ void pack_w_kernel(const float* __restrict__ W,
                              unsigned* __restrict__ Wp) {
    int i = blockIdx.x * blockDim.x + threadIdx.x;
    if (i >= 2048) return;
    int v    = i & 7;
    int lane = (i >> 3) & 31;
    int w    = (i >> 8) & 3;
    int s    = i >> 10;
    int half = lane >> 4, l15 = lane & 15;
    int col  = w * 16 + l15;
    int k    = 32 * s + 16 * half + 2 * v;
    Wp[i] = pack_bf16x2(W[k * 64 + col], W[(k + 1) * 64 + col]);
}

// ---------------------------------------------------------------------------
// Self-loop attr: mean of incoming edge_attr per dst (fill_value='mean')
// ---------------------------------------------------------------------------
__global__ void loop_scatter(const int* __restrict__ ei,
                             const float* __restrict__ eattr,
                             float* __restrict__ lsum,
                             float* __restrict__ cnt, int E) {
    int t = blockIdx.x * blockDim.x + threadIdx.x;
    if (t >= E * 16) return;
    int e = t >> 4, c = t & 15;
    int d = ei[E + e];
    atomicAdd(&lsum[(size_t)d * 16 + c], eattr[t]);
    if (c == 0) atomicAdd(&cnt[d], 1.0f);
}

__global__ void loop_divide(float* __restrict__ lsum,
                            const float* __restrict__ cnt, int N) {
    int t = blockIdx.x * blockDim.x + threadIdx.x;
    if (t >= N * 16) return;
    lsum[t] = lsum[t] / fmaxf(cnt[t >> 4], 1.0f);
}

// ---------------------------------------------------------------------------
// Y[nrows,64] = X[nrows,64] @ W[64,64] + bias.
// Block = 128 threads (4 waves); block tile 64 rows x 64 cols.
// Wave w owns rows [row0+16w, +16) x all 4 column tiles -> 8 WMMA per wave.
// A fragments staged pre-packed in LDS (2x ds_load_b128 per fragment);
// B fragments read pre-packed from global (2x global_load_b128, L2-resident).
// ---------------------------------------------------------------------------
__global__ void gemm64_wmma(const float* __restrict__ X,
                            const v8u* __restrict__ Wp,     // [2][4][32] packed
                            const float* __restrict__ bias, // [64]
                            float* __restrict__ Y, int nrows) {
    __shared__ v8u Asp[4][2][32];   // [rowtile][kstep][lane] = 8KB

    const int tid  = threadIdx.x;   // 0..127
    const int wave = tid >> 5;
    const int lane = tid & 31;
    const int row0 = blockIdx.x * 64;

    // ---- stage A fragments: 2048 packed dwords by 128 threads ----
    unsigned* AspU = (unsigned*)Asp;
    for (int i = tid; i < 2048; i += 128) {
        int v   = i & 7;
        int ln  = (i >> 3) & 31;
        int s   = (i >> 8) & 1;
        int rt  = i >> 9;
        int half = ln >> 4, l15 = ln & 15;
        // A 16x32 bf16 map: dword v -> k = 16*half + 2v (matches packed B)
        int ka  = half * 16 + 2 * v;
        int row = row0 + rt * 16 + l15;
        float lo = 0.0f, hi = 0.0f;
        if (row < nrows) {
            const float* px = X + (size_t)row * 64 + 32 * s + ka;
            lo = px[0]; hi = px[1];
        }
        AspU[i] = pack_bf16x2(lo, hi);
    }
    __syncthreads();

    const int half = lane >> 4;
    const int l15  = lane & 15;

    v16bf a0 = __builtin_bit_cast(v16bf, Asp[wave][0][lane]);
    v16bf a1 = __builtin_bit_cast(v16bf, Asp[wave][1][lane]);

    const int rowbase = row0 + wave * 16;
    const bool fullTile = (row0 + 64) <= nrows;

#pragma unroll
    for (int c = 0; c < 4; ++c) {
        v16bf b0 = __builtin_bit_cast(v16bf, Wp[(0 * 4 + c) * 32 + lane]);
        v16bf b1 = __builtin_bit_cast(v16bf, Wp[(1 * 4 + c) * 32 + lane]);
        v8f acc = {};
        acc = __builtin_amdgcn_wmma_f32_16x16x32_bf16(
                  false, a0, false, b0, (short)0, acc, false, false);
        acc = __builtin_amdgcn_wmma_f32_16x16x32_bf16(
                  false, a1, false, b1, (short)0, acc, false, false);

        const int col = c * 16 + l15;
        const float bv = bias[col];
        if (fullTile) {
#pragma unroll
            for (int r = 0; r < 8; ++r)
                Y[(size_t)(rowbase + r + half * 8) * 64 + col] = acc[r] + bv;
        } else {
#pragma unroll
            for (int r = 0; r < 8; ++r) {
                int row = rowbase + r + half * 8;
                if (row < nrows) Y[(size_t)row * 64 + col] = acc[r] + bv;
            }
        }
    }
}

// ---------------------------------------------------------------------------
// Pass A: per-edge attention logit with fused ee = ea @ We.
// One wave per edge; lane covers h and h+32. Wave-reduce -> escore,
// atomic segment-max via monotonic uint keys.
// ---------------------------------------------------------------------------
__global__ void edge_score(const float* __restrict__ xl,
                           const float* __restrict__ xr,
                           const float* __restrict__ We,       // [16][64]
                           const float* __restrict__ att,      // [64]
                           const int* __restrict__ ei,         // [2*E]
                           const float* __restrict__ eattr,    // [E*16]
                           const float* __restrict__ loop_attr,// [N*16]
                           float* __restrict__ escore,
                           unsigned* __restrict__ emax_u,
                           int E, int N) {
    __shared__ float sWe[16 * 64];
    for (int i = threadIdx.x; i < 16 * 64; i += blockDim.x) sWe[i] = We[i];
    __syncthreads();

    int wid  = (int)(((size_t)blockIdx.x * blockDim.x + threadIdx.x) >> 5);
    int lane = threadIdx.x & 31;
    if (wid >= E + N) return;

    int s, d;
    const float* ea;
    if (wid < E) { s = ei[wid]; d = ei[E + wid]; ea = eattr + (size_t)wid * 16; }
    else         { s = d = wid - E;              ea = loop_attr + (size_t)(wid - E) * 16; }

    float eav = (lane < 16) ? ea[lane] : 0.0f;

    float total = 0.0f;
#pragma unroll
    for (int p = 0; p < 2; ++p) {
        int   h = lane + 32 * p;
        float v = xl[(size_t)s * 64 + h] + xr[(size_t)d * 64 + h];
#pragma unroll
        for (int k = 0; k < 16; ++k) {
            float ak = __shfl(eav, k, 32);
            v = fmaf(ak, sWe[k * 64 + h], v);
        }
        v = (v > 0.0f) ? v : NEG_SLOPE * v;   // leaky_relu
        total = fmaf(v, att[h], total);
    }
#pragma unroll
    for (int off = 16; off > 0; off >>= 1)
        total += __shfl_xor(total, off, 32);

    if (lane == 0) {
        escore[wid] = total;
        atomicMax(&emax_u[d], fkey(total));
    }
}

// Pass B: a = exp(e - emax[dst]); denom[dst] += a
__global__ void edge_exp(const float* __restrict__ escore,
                         const unsigned* __restrict__ emax_u,
                         const int* __restrict__ ei,
                         float* __restrict__ aexp,
                         float* __restrict__ denom, int E, int N) {
    int i = blockIdx.x * blockDim.x + threadIdx.x;
    if (i >= E + N) return;
    int d = (i < E) ? ei[E + i] : (i - E);
    float m = funkey(emax_u[d]);
    if (!__builtin_isfinite(m)) m = 0.0f;
    float a = __expf(escore[i] - m);
    aexp[i] = a;
    atomicAdd(&denom[d], a);
}

// Pass C: agg[dst] += alpha * xl[src]   (wave per edge; lane -> h, h+32)
__global__ void edge_aggregate(const float* __restrict__ xl,
                               const int* __restrict__ ei,
                               const float* __restrict__ aexp,
                               const float* __restrict__ denom,
                               float* __restrict__ agg, int E, int N) {
    int wid  = (int)(((size_t)blockIdx.x * blockDim.x + threadIdx.x) >> 5);
    int lane = threadIdx.x & 31;
    if (wid >= E + N) return;
    int s, d;
    if (wid < E) { s = ei[wid]; d = ei[E + wid]; }
    else         { s = d = wid - E; }
    float alpha = aexp[wid] / (denom[d] + EPS_DEN);
    atomicAdd(&agg[(size_t)d * 64 + lane],      alpha * xl[(size_t)s * 64 + lane]);
    atomicAdd(&agg[(size_t)d * 64 + lane + 32], alpha * xl[(size_t)s * 64 + lane + 32]);
}

// Pass D: h = relu(agg + bc)
__global__ void finalize_node(const float* __restrict__ agg,
                              const float* __restrict__ bc,
                              float* __restrict__ hout, int N) {
    int i = blockIdx.x * blockDim.x + threadIdx.x;
    if (i >= N * 64) return;
    float v = agg[i] + bc[i & 63];
    hout[i] = v > 0.0f ? v : 0.0f;
}

// JK-cat readout: y[n] = [h1|h2] @ Wout + bout   (wave per node)
__global__ void readout(const float* __restrict__ h1,
                        const float* __restrict__ h2,
                        const float* __restrict__ Wout,  // [128]
                        const float* __restrict__ bout,  // [1]
                        float* __restrict__ y, int N) {
    int wid  = (int)(((size_t)blockIdx.x * blockDim.x + threadIdx.x) >> 5);
    int lane = threadIdx.x & 31;
    if (wid >= N) return;
    size_t b = (size_t)wid * 64;
    float t = h1[b + lane]      * Wout[lane]
            + h1[b + lane + 32] * Wout[lane + 32]
            + h2[b + lane]      * Wout[64 + lane]
            + h2[b + lane + 32] * Wout[96 + lane];
#pragma unroll
    for (int off = 16; off > 0; off >>= 1)
        t += __shfl_xor(t, off, 32);
    if (lane == 0) y[wid] = t + bout[0];
}

// ---------------------------------------------------------------------------
extern "C" void kernel_launch(void* const* d_in, const int* in_sizes, int n_in,
                              void* d_out, int out_size, void* d_ws, size_t ws_size,
                              hipStream_t stream) {
    const float* x     = (const float*)d_in[0];
    const int*   ei    = (const int*)d_in[1];
    const float* eattr = (const float*)d_in[2];
    // d_in[3] = batch (unused; single graph)
    const float *Wl1 = (const float*)d_in[4],  *bl1 = (const float*)d_in[5];
    const float *Wr1 = (const float*)d_in[6],  *br1 = (const float*)d_in[7];
    const float *We1 = (const float*)d_in[8],  *att1 = (const float*)d_in[9];
    const float *bc1 = (const float*)d_in[10];
    const float *Wl2 = (const float*)d_in[11], *bl2 = (const float*)d_in[12];
    const float *Wr2 = (const float*)d_in[13], *br2 = (const float*)d_in[14];
    const float *We2 = (const float*)d_in[15], *att2 = (const float*)d_in[16];
    const float *bc2 = (const float*)d_in[17];
    const float *Wout = (const float*)d_in[18], *bout = (const float*)d_in[19];

    const int N = in_sizes[0] / 64;
    const int E = in_sizes[2] / 16;
    const int EN = E + N;

    // ---- workspace carve (bump allocator, 256B aligned) ----
    char* base = (char*)d_ws;
    size_t off = 0;
    auto alloc = [&](size_t bytes) -> void* {
        void* p = base + off;
        off = (off + bytes + 255) & ~(size_t)255;
        return p;
    };
    float*    loop_attr = (float*)alloc((size_t)N * 16 * 4);  // sum -> mean in place
    float*    cnt       = (float*)alloc((size_t)N * 4);
    float*    xl        = (float*)alloc((size_t)N * 64 * 4);
    float*    xr        = (float*)alloc((size_t)N * 64 * 4);
    float*    h1        = (float*)alloc((size_t)N * 64 * 4);
    float*    h2        = (float*)alloc((size_t)N * 64 * 4);
    float*    agg       = (float*)alloc((size_t)N * 64 * 4);
    float*    denom     = (float*)alloc((size_t)N * 4);
    unsigned* emax_u    = (unsigned*)alloc((size_t)N * 4);
    float*    escore    = (float*)alloc((size_t)EN * 4);
    float*    aexp      = (float*)alloc((size_t)EN * 4);
    unsigned* Wl1p      = (unsigned*)alloc(2048 * 4);   // packed B fragments
    unsigned* Wr1p      = (unsigned*)alloc(2048 * 4);
    unsigned* Wl2p      = (unsigned*)alloc(2048 * 4);
    unsigned* Wr2p      = (unsigned*)alloc(2048 * 4);

    const int T = 256;
    auto blocks = [&](size_t n) { return (unsigned)((n + T - 1) / T); };

    // ---- self-loop mean edge_attr ----
    fill_u32<<<blocks((size_t)N * 16), T, 0, stream>>>((unsigned*)loop_attr, 0u, (size_t)N * 16);
    fill_u32<<<blocks(N), T, 0, stream>>>((unsigned*)cnt, 0u, (size_t)N);
    loop_scatter<<<blocks((size_t)E * 16), T, 0, stream>>>(ei, eattr, loop_attr, cnt, E);
    loop_divide<<<blocks((size_t)N * 16), T, 0, stream>>>(loop_attr, cnt, N);

    // ---- pack weights into WMMA B-fragment layout (bf16) ----
    pack_w_kernel<<<8, T, 0, stream>>>(Wl1, Wl1p);
    pack_w_kernel<<<8, T, 0, stream>>>(Wr1, Wr1p);
    pack_w_kernel<<<8, T, 0, stream>>>(Wl2, Wl2p);
    pack_w_kernel<<<8, T, 0, stream>>>(Wr2, Wr2p);

    const unsigned gemmBlocks = (unsigned)((N + 63) / 64);
    const unsigned waveBlocks = blocks((size_t)EN * 32);

    // =============== layer 1 ===============
    gemm64_wmma<<<gemmBlocks, 128, 0, stream>>>(x, (const v8u*)Wl1p, bl1, xl, N);
    gemm64_wmma<<<gemmBlocks, 128, 0, stream>>>(x, (const v8u*)Wr1p, br1, xr, N);

    fill_u32<<<blocks(N), T, 0, stream>>>(emax_u, KEY_NEGMAX, (size_t)N);
    fill_u32<<<blocks(N), T, 0, stream>>>((unsigned*)denom, 0u, (size_t)N);
    fill_u32<<<blocks((size_t)N * 64), T, 0, stream>>>((unsigned*)agg, 0u, (size_t)N * 64);

    edge_score<<<waveBlocks, T, 0, stream>>>(xl, xr, We1, att1, ei, eattr, loop_attr,
                                             escore, emax_u, E, N);
    edge_exp<<<blocks(EN), T, 0, stream>>>(escore, emax_u, ei, aexp, denom, E, N);
    edge_aggregate<<<waveBlocks, T, 0, stream>>>(xl, ei, aexp, denom, agg, E, N);
    finalize_node<<<blocks((size_t)N * 64), T, 0, stream>>>(agg, bc1, h1, N);

    // =============== layer 2 ===============
    gemm64_wmma<<<gemmBlocks, 128, 0, stream>>>(h1, (const v8u*)Wl2p, bl2, xl, N);
    gemm64_wmma<<<gemmBlocks, 128, 0, stream>>>(h1, (const v8u*)Wr2p, br2, xr, N);

    fill_u32<<<blocks(N), T, 0, stream>>>(emax_u, KEY_NEGMAX, (size_t)N);
    fill_u32<<<blocks(N), T, 0, stream>>>((unsigned*)denom, 0u, (size_t)N);
    fill_u32<<<blocks((size_t)N * 64), T, 0, stream>>>((unsigned*)agg, 0u, (size_t)N * 64);

    edge_score<<<waveBlocks, T, 0, stream>>>(xl, xr, We2, att2, ei, eattr, loop_attr,
                                             escore, emax_u, E, N);
    edge_exp<<<blocks(EN), T, 0, stream>>>(escore, emax_u, ei, aexp, denom, E, N);
    edge_aggregate<<<waveBlocks, T, 0, stream>>>(xl, ei, aexp, denom, agg, E, N);
    finalize_node<<<blocks((size_t)N * 64), T, 0, stream>>>(agg, bc2, h2, N);

    // =============== readout ===============
    readout<<<blocks((size_t)N * 32), T, 0, stream>>>(h1, h2, Wout, bout, (float*)d_out, N);
}
